// SwinBottleneck_56367150792964
// MI455X (gfx1250) — compile-verified, matrix operand order checked by reading, not run on hardware
//
#include <hip/hip_runtime.h>
#include <hip/hip_bf16.h>
#include <math.h>

// ---------------------------------------------------------------------------
// Problem constants (from the reference)
// ---------------------------------------------------------------------------
#define BB 8
#define CC 320
#define DD 32
#define HH 32
#define WWID 16
#define HEADS 10
#define HDIM 32
#define LL (DD * HH * WWID)          // 16384
#define NTOK 32                      // tokens per window
#define NWIN 512                     // windows per sample
#define MTOT (BB * LL)               // 131072 rows
#define GELU_C 0.70710678118654752f

typedef __attribute__((ext_vector_type(16))) __bf16 v16bf;
typedef __attribute__((ext_vector_type(8)))  float  v8f;

union FragAB { uint4 u4[2]; v16bf v; };
union AccU   { v8f v; float f[8]; };

static __device__ __forceinline__ unsigned short f2bf(float f) {
    unsigned u = __float_as_uint(f);
    unsigned r = (u + 0x7FFFu + ((u >> 16) & 1u)) >> 16;
    return (unsigned short)r;
}

// Low 32 bits of a generic pointer to __shared__ = LDS byte offset (aperture
// layout: addr[31:0] is the LDS address).
static __device__ __forceinline__ unsigned lds_addr_of(const void* p) {
    return (unsigned)(unsigned long long)(uintptr_t)p;
}

// CDNA5 async global->LDS copy (16 bytes per lane), tracked by ASYNCcnt.
static __device__ __forceinline__ void async_load_b128(unsigned lds, const void* gaddr) {
    asm volatile("global_load_async_to_lds_b128 %0, %1, off"
                 :: "v"(lds), "v"((unsigned long long)(uintptr_t)gaddr)
                 : "memory");
}
static __device__ __forceinline__ void wait_async0() {
    asm volatile("s_wait_asynccnt 0x0" ::: "memory");
}

// ---------------------------------------------------------------------------
// Tiled transposes between (B,C,L) and (B,L,C); coalesced both ways via LDS.
// DIR 0: x(B,C,L) -> t(B,L,C)      DIR 1: t(B,L,C) -> out(B,C,L)
// ---------------------------------------------------------------------------
template <int DIR>
__global__ __launch_bounds__(256) void transpose_kernel(const float* __restrict__ in,
                                                        float* __restrict__ out) {
    __shared__ float tile[32][33];
    int gid = blockIdx.x;                 // B * (CC/32) * (LL/32) = 8*10*512
    int b   = gid / 5120;
    int rem = gid - b * 5120;
    int tc  = rem >> 9;                   // 0..9
    int tl  = rem & 511;                  // 0..511
    int c0 = tc * 32, l0 = tl * 32;
    int tx = threadIdx.x & 31;
    int ty = threadIdx.x >> 5;            // 0..7

#pragma unroll
    for (int j = 0; j < 4; ++j) {
        int r = ty + 8 * j;
        if (DIR == 0)
            tile[r][tx] = in[((size_t)(b * CC + c0 + r)) * LL + l0 + tx];
        else
            tile[r][tx] = in[((size_t)(b * LL + l0 + r)) * CC + c0 + tx];
    }
    __syncthreads();
#pragma unroll
    for (int j = 0; j < 4; ++j) {
        int r = ty + 8 * j;
        if (DIR == 0)
            out[((size_t)(b * LL + l0 + r)) * CC + c0 + tx] = tile[tx][r];
        else
            out[((size_t)(b * CC + c0 + r)) * LL + l0 + tx] = tile[tx][r];
    }
}

// weight (K x N) fp32 row-major  ->  (N x K) bf16 row-major
__global__ __launch_bounds__(256) void wt_kernel(const float* __restrict__ in,
                                                 unsigned short* __restrict__ out,
                                                 int K, int N) {
    size_t gid = (size_t)blockIdx.x * 256 + threadIdx.x;      // over N*K
    int k = (int)(gid % K);
    int n = (int)(gid / K);
    out[gid] = f2bf(in[(size_t)k * N + n]);
}

// bias (147, HEADS) -> biasNN[head][n][m]  (HEADS x 32 x 32 fp32)
__global__ __launch_bounds__(1024) void bias_nn_kernel(const float* __restrict__ tab,
                                                       float* __restrict__ out) {
    int tid = threadIdx.x;                 // 1024 = n*32 + m
    int n = tid >> 5, m = tid & 31;
    int dd = (n >> 3) - (m >> 3) + 3;
    int dh = ((n >> 1) & 3) - ((m >> 1) & 3) + 3;
    int dw = (n & 1) - (m & 1) + 1;
    int idx = dd * 21 + dh * 3 + dw;
    for (int h = 0; h < HEADS; ++h)
        out[h * 1024 + tid] = tab[idx * HEADS + h];
}

// shift-mask region ids: wid[wloc][n]
__global__ __launch_bounds__(256) void wid_kernel(int* __restrict__ wid) {
    int tid = blockIdx.x * 256 + threadIdx.x;   // 512*32
    int wloc = tid >> 5, n = tid & 31;
    int dB = wloc >> 6, hB = (wloc >> 3) & 7, wB = wloc & 7;
    int d = dB * 4 + (n >> 3);
    int h = hB * 4 + ((n >> 1) & 3);
    int w = wB * 2 + (n & 1);
    int rd = (d < 28) ? 0 : ((d < 30) ? 1 : 2);
    int rh = (h < 28) ? 0 : ((h < 30) ? 1 : 2);
    int rw = (w < 14) ? 0 : ((w < 15) ? 1 : 2);
    wid[tid] = (rd * 3 + rh) * 3 + rw;
}

// ---------------------------------------------------------------------------
// LayerNorm over C=320, one wave32 per token.
// WMODE: 0 = plain row order; 1 = window partition; 2 = shift + window part.
// ---------------------------------------------------------------------------
template <int WMODE>
__global__ __launch_bounds__(256) void ln_kernel(const float* __restrict__ t,
                                                 const float* __restrict__ gam,
                                                 const float* __restrict__ bet,
                                                 unsigned short* __restrict__ dst) {
    int wave = threadIdx.x >> 5;
    int lane = threadIdx.x & 31;
    int token = blockIdx.x * 8 + wave;
    size_t base = (size_t)token * CC;

    float v[10];
    float sum = 0.f, sq = 0.f;
#pragma unroll
    for (int j = 0; j < 10; ++j) {
        v[j] = t[base + lane + 32 * j];
        sum += v[j];
        sq  += v[j] * v[j];
    }
#pragma unroll
    for (int off = 16; off > 0; off >>= 1) {
        sum += __shfl_xor(sum, off, 32);
        sq  += __shfl_xor(sq,  off, 32);
    }
    float mean = sum * (1.f / CC);
    float var  = sq * (1.f / CC) - mean * mean;
    float rstd = rsqrtf(var + 1e-5f);

    int b = token >> 14;
    int l = token & (LL - 1);
    int row;
    if (WMODE == 0) {
        row = token;
    } else {
        int d = l >> 9, h = (l >> 4) & 31, w = l & 15;
        if (WMODE == 2) { d = (d + 30) & 31; h = (h + 30) & 31; w = (w + 15) & 15; }
        int dB = d >> 2, di = d & 3;
        int hB = h >> 2, hi = h & 3;
        int wB = w >> 1, wi = w & 1;
        row = (b * NWIN + dB * 64 + hB * 8 + wB) * NTOK + (di * 8 + hi * 2 + wi);
    }
    unsigned short* o = dst + (size_t)row * CC;
#pragma unroll
    for (int j = 0; j < 10; ++j) {
        int c = lane + 32 * j;
        o[c] = f2bf((v[j] - mean) * rstd * gam[c] + bet[c]);
    }
}

// ---------------------------------------------------------------------------
// Tiled bf16 WMMA GEMM:  D = A(MxK) * Bt^T  (Bt stored as Nc x K) + bias.
// 256 threads = 8 waves -> 128x64 tile; wave = 32x32 (2x2 WMMA accumulators).
// Double-buffered LDS fed by GLOBAL_LOAD_ASYNC_TO_LDS_B128 (ASYNCcnt),
// overlapping the next tile's HBM->LDS DMA with the current tile's WMMAs.
// MODE 0: QKV scatter (q scaled, v transposed)   MODE 1: proj + residual(window)
// MODE 2: GELU -> bf16 hidden                    MODE 3: residual (plain order)
// ---------------------------------------------------------------------------
template <int MODE, int SHIFTED>
__global__ __launch_bounds__(256) void gemm_kernel(const unsigned short* __restrict__ A,
                                                   const unsigned short* __restrict__ Bt,
                                                   const float* __restrict__ bias,
                                                   int M, int Nc, int K,
                                                   unsigned short* __restrict__ outA,
                                                   unsigned short* __restrict__ outB,
                                                   unsigned short* __restrict__ outC,
                                                   float* __restrict__ tRes) {
    __shared__ unsigned short Als[2][128 * 32];
    __shared__ unsigned short Bls[2][64 * 32];

    int tid = threadIdx.x;
    int lane = tid & 31;
    int wave = tid >> 5;
    int wm = wave & 3;          // 0..3 -> 32-row chunk
    int wn = wave >> 2;         // 0..1 -> 32-col chunk
    int rowBlk = blockIdx.x * 128;
    int colBlk = blockIdx.y * 64;
    int lr   = lane & 15;
    int koff = (lane >> 4) * 8;
    int nsteps = K >> 5;

    // issue the async HBM->LDS copies for K-step `step` into LDS buffer `buf`
    auto issue = [&](int step, int buf) {
        int kk = step << 5;
#pragma unroll
        for (int e = tid; e < 512; e += 256) {          // A: 128x32 bf16
            int r = e >> 2, s = e & 3;
            async_load_b128(lds_addr_of(&Als[buf][r * 32 + s * 8]),
                            &A[(size_t)(rowBlk + r) * K + kk + s * 8]);
        }
        {                                               // B: 64x32 bf16
            int r = tid >> 2, s = tid & 3;
            async_load_b128(lds_addr_of(&Bls[buf][r * 32 + s * 8]),
                            &Bt[(size_t)(colBlk + r) * K + kk + s * 8]);
        }
    };

    v8f acc[2][2] = {};

    issue(0, 0);
    for (int step = 0; step < nsteps; ++step) {
        int buf = step & 1;
        wait_async0();          // our async writes into `buf` are in LDS
        __syncthreads();        // everyone's writes visible; prev reads done
        if (step + 1 < nsteps) issue(step + 1, buf ^ 1);

        FragAB a[2], b[2];
#pragma unroll
        for (int i = 0; i < 2; ++i) {
            int m = wm * 32 + i * 16 + lr;
            a[i].u4[0] = *(const uint4*)&Als[buf][m * 32 + koff];
            a[i].u4[1] = *(const uint4*)&Als[buf][m * 32 + 16 + koff];
        }
#pragma unroll
        for (int j = 0; j < 2; ++j) {
            int n = wn * 32 + j * 16 + lr;
            b[j].u4[0] = *(const uint4*)&Bls[buf][n * 32 + koff];
            b[j].u4[1] = *(const uint4*)&Bls[buf][n * 32 + 16 + koff];
        }
#pragma unroll
        for (int i = 0; i < 2; ++i)
#pragma unroll
            for (int j = 0; j < 2; ++j)
                acc[i][j] = __builtin_amdgcn_wmma_f32_16x16x32_bf16(
                    false, a[i].v, false, b[j].v, (short)0, acc[i][j], false, false);
    }

    // Epilogue
#pragma unroll
    for (int i = 0; i < 2; ++i)
#pragma unroll
        for (int j = 0; j < 2; ++j) {
            AccU u; u.v = acc[i][j];
            int gcol = colBlk + wn * 32 + j * 16 + lr;
            float bv = bias[gcol];
#pragma unroll
            for (int r = 0; r < 8; ++r) {
                int grow = rowBlk + wm * 32 + i * 16 + r + ((lane >> 4) << 3);
                float val = u.f[r] + bv;
                if (MODE == 0) {
                    int sel = gcol / CC;
                    int rem = gcol - sel * CC;
                    int head = rem >> 5, dd = rem & 31;
                    int win = grow >> 5, n = grow & 31;
                    size_t base = (size_t)(win * HEADS + head) * 1024;
                    if (sel == 0)      outA[base + n * 32 + dd] = f2bf(val * 0.17677669529663689f);
                    else if (sel == 1) outB[base + n * 32 + dd] = f2bf(val);
                    else               outC[base + dd * 32 + n] = f2bf(val);  // V transposed
                } else if (MODE == 1) {
                    int win = grow >> 5, n = grow & 31;
                    int b = win >> 9, wloc = win & 511;
                    int d = (wloc >> 6) * 4 + (n >> 3);
                    int h = ((wloc >> 3) & 7) * 4 + ((n >> 1) & 3);
                    int w = (wloc & 7) * 2 + (n & 1);
                    if (SHIFTED) { d = (d + 2) & 31; h = (h + 2) & 31; w = (w + 1) & 15; }
                    size_t idx = ((size_t)(b * LL) + ((d << 9) | (h << 4) | w)) * CC + gcol;
                    tRes[idx] = tRes[idx] + val;
                } else if (MODE == 2) {
                    float g = 0.5f * val * (1.f + erff(val * GELU_C));
                    outA[(size_t)grow * Nc + gcol] = f2bf(g);
                } else { // MODE == 3
                    size_t idx = (size_t)grow * CC + gcol;
                    tRes[idx] = tRes[idx] + val;
                }
            }
        }
}

// ---------------------------------------------------------------------------
// Attention: one wave32 per (window, head). QK^T -> softmax -> P*V, all WMMA.
// ---------------------------------------------------------------------------
__global__ __launch_bounds__(256) void attn_kernel(const unsigned short* __restrict__ q,
                                                   const unsigned short* __restrict__ k,
                                                   const unsigned short* __restrict__ vt,
                                                   const float* __restrict__ biasNN,
                                                   const int* __restrict__ wid,
                                                   unsigned short* __restrict__ attnOut,
                                                   int shifted) {
    __shared__ float          Ssh[8][1024];
    __shared__ unsigned short Psh[8][1024];

    int tid = threadIdx.x;
    int wave = tid >> 5;
    int lane = tid & 31;
    int unit = blockIdx.x * 8 + wave;         // (win*HEADS + head)
    int win  = unit / HEADS;
    int head = unit - win * HEADS;
    int lr   = lane & 15;
    int koff = (lane >> 4) * 8;

    const unsigned short* qb = q  + (size_t)unit * 1024;   // [n][d]
    const unsigned short* kb = k  + (size_t)unit * 1024;   // [m][d]
    const unsigned short* vb = vt + (size_t)unit * 1024;   // [d][m]

    // ---- S = (Q*scale) K^T ----
    FragAB a[2], b[2];
#pragma unroll
    for (int i = 0; i < 2; ++i) {
        int m = i * 16 + lr;
        a[i].u4[0] = *(const uint4*)&qb[m * 32 + koff];
        a[i].u4[1] = *(const uint4*)&qb[m * 32 + 16 + koff];
    }
#pragma unroll
    for (int j = 0; j < 2; ++j) {
        int n = j * 16 + lr;
        b[j].u4[0] = *(const uint4*)&kb[n * 32 + koff];
        b[j].u4[1] = *(const uint4*)&kb[n * 32 + 16 + koff];
    }
    v8f s[2][2] = {};
#pragma unroll
    for (int i = 0; i < 2; ++i)
#pragma unroll
        for (int j = 0; j < 2; ++j)
            s[i][j] = __builtin_amdgcn_wmma_f32_16x16x32_bf16(
                false, a[i].v, false, b[j].v, (short)0, s[i][j], false, false);
#pragma unroll
    for (int i = 0; i < 2; ++i)
#pragma unroll
        for (int j = 0; j < 2; ++j) {
            AccU u; u.v = s[i][j];
#pragma unroll
            for (int r = 0; r < 8; ++r) {
                int row = i * 16 + r + ((lane >> 4) << 3);
                Ssh[wave][row * 32 + j * 16 + lr] = u.f[r];
            }
        }
    __syncthreads();

    // ---- softmax: one lane per row ----
    {
        int n = lane;
        const float* bNN = biasNN + head * 1024 + n * 32;
        const int*   wl  = wid + (win & 511) * 32;
        int myid = wl[n];
        float sv[32];
        float mx = -1e30f;
#pragma unroll
        for (int m = 0; m < 32; ++m) {
            float sc = Ssh[wave][n * 32 + m] + bNN[m];
            if (shifted && wl[m] != myid) sc -= 100.f;
            sv[m] = sc;
            mx = fmaxf(mx, sc);
        }
        float sum = 0.f;
#pragma unroll
        for (int m = 0; m < 32; ++m) { sv[m] = __expf(sv[m] - mx); sum += sv[m]; }
        float inv = 1.f / sum;
#pragma unroll
        for (int m = 0; m < 32; ++m)
            Psh[wave][n * 32 + m] = f2bf(sv[m] * inv);
    }
    __syncthreads();

    // ---- O = P * V ----
#pragma unroll
    for (int i = 0; i < 2; ++i) {
        int n = i * 16 + lr;
        a[i].u4[0] = *(const uint4*)&Psh[wave][n * 32 + koff];
        a[i].u4[1] = *(const uint4*)&Psh[wave][n * 32 + 16 + koff];
    }
#pragma unroll
    for (int j = 0; j < 2; ++j) {
        int d = j * 16 + lr;
        b[j].u4[0] = *(const uint4*)&vb[d * 32 + koff];
        b[j].u4[1] = *(const uint4*)&vb[d * 32 + 16 + koff];
    }
    v8f o[2][2] = {};
#pragma unroll
    for (int i = 0; i < 2; ++i)
#pragma unroll
        for (int j = 0; j < 2; ++j)
            o[i][j] = __builtin_amdgcn_wmma_f32_16x16x32_bf16(
                false, a[i].v, false, b[j].v, (short)0, o[i][j], false, false);
#pragma unroll
    for (int i = 0; i < 2; ++i)
#pragma unroll
        for (int j = 0; j < 2; ++j) {
            AccU u; u.v = o[i][j];
#pragma unroll
            for (int r = 0; r < 8; ++r) {
                int n = i * 16 + r + ((lane >> 4) << 3);
                int d = j * 16 + lr;
                attnOut[(size_t)(win * 32 + n) * CC + head * 32 + d] = f2bf(u.f[r]);
            }
        }
}

// ---------------------------------------------------------------------------
// Host side
// ---------------------------------------------------------------------------
static inline size_t alignUp(size_t x) { return (x + 255) & ~(size_t)255; }

extern "C" void kernel_launch(void* const* d_in, const int* in_sizes, int n_in,
                              void* d_out, int out_size, void* d_ws, size_t ws_size,
                              hipStream_t stream) {
    (void)in_sizes; (void)n_in; (void)out_size; (void)ws_size;

    const float* x = (const float*)d_in[0];
    // per-block params: ln1g ln1b qkvw qkvb bias projw projb ln2g ln2b f1w f1b f2w f2b
    const float* P[2][13];
    for (int blk = 0; blk < 2; ++blk)
        for (int j = 0; j < 13; ++j)
            P[blk][j] = (const float*)d_in[1 + blk * 13 + j];

    // ---- carve workspace ----
    char* ws = (char*)d_ws;
    size_t off = 0;
    auto carve = [&](size_t bytes) { char* p = ws + off; off += alignUp(bytes); return (void*)p; };

    float*          t      = (float*)carve((size_t)MTOT * CC * 4);          // residual stream
    unsigned short* xw     = (unsigned short*)carve((size_t)MTOT * CC * 2); // LN1 out / attnOut
    unsigned short* qb     = (unsigned short*)carve((size_t)MTOT * CC * 2); // Q  / LN2 out
    unsigned short* kb     = (unsigned short*)carve((size_t)MTOT * CC * 2); // K
    unsigned short* vb     = (unsigned short*)carve((size_t)MTOT * CC * 2); // V^T
    unsigned short* hidden = (unsigned short*)carve((size_t)MTOT * 4 * CC * 2);
    unsigned short* wtQ[2], *wtP[2], *wtF1[2], *wtF2[2];
    float* biasNN[2];
    for (int blk = 0; blk < 2; ++blk) {
        wtQ[blk]    = (unsigned short*)carve((size_t)960 * 320 * 2);
        wtP[blk]    = (unsigned short*)carve((size_t)320 * 320 * 2);
        wtF1[blk]   = (unsigned short*)carve((size_t)1280 * 320 * 2);
        wtF2[blk]   = (unsigned short*)carve((size_t)320 * 1280 * 2);
        biasNN[blk] = (float*)carve((size_t)HEADS * 1024 * 4);
    }
    int* wid = (int*)carve((size_t)NWIN * NTOK * 4);

    // ---- prep ----
    transpose_kernel<0><<<BB * 10 * 512, 256, 0, stream>>>(x, t);
    for (int blk = 0; blk < 2; ++blk) {
        wt_kernel<<<(960 * 320) / 256, 256, 0, stream>>>(P[blk][2], wtQ[blk], 320, 960);
        wt_kernel<<<(320 * 320) / 256, 256, 0, stream>>>(P[blk][5], wtP[blk], 320, 320);
        wt_kernel<<<(1280 * 320) / 256, 256, 0, stream>>>(P[blk][9], wtF1[blk], 320, 1280);
        wt_kernel<<<(320 * 1280) / 256, 256, 0, stream>>>(P[blk][11], wtF2[blk], 1280, 320);
        bias_nn_kernel<<<1, 1024, 0, stream>>>(P[blk][4], biasNN[blk]);
    }
    wid_kernel<<<(NWIN * NTOK) / 256, 256, 0, stream>>>(wid);

    dim3 blkT(256);
    dim3 gQKV(MTOT / 128, 960 / 64);
    dim3 gPRJ(MTOT / 128, 320 / 64);
    dim3 gF1(MTOT / 128, 1280 / 64);
    dim3 gF2(MTOT / 128, 320 / 64);

    // ================= block 1 (not shifted) =================
    ln_kernel<1><<<MTOT / 8, 256, 0, stream>>>(t, P[0][0], P[0][1], xw);
    gemm_kernel<0, 0><<<gQKV, blkT, 0, stream>>>(xw, wtQ[0], P[0][3], MTOT, 960, 320,
                                                 qb, kb, vb, nullptr);
    attn_kernel<<<(MTOT / NTOK * HEADS) / 8, 256, 0, stream>>>(qb, kb, vb, biasNN[0], wid, xw, 0);
    gemm_kernel<1, 0><<<gPRJ, blkT, 0, stream>>>(xw, wtP[0], P[0][6], MTOT, 320, 320,
                                                 nullptr, nullptr, nullptr, t);
    ln_kernel<0><<<MTOT / 8, 256, 0, stream>>>(t, P[0][7], P[0][8], qb);
    gemm_kernel<2, 0><<<gF1, blkT, 0, stream>>>(qb, wtF1[0], P[0][10], MTOT, 1280, 320,
                                                hidden, nullptr, nullptr, nullptr);
    gemm_kernel<3, 0><<<gF2, blkT, 0, stream>>>(hidden, wtF2[0], P[0][12], MTOT, 320, 1280,
                                                nullptr, nullptr, nullptr, t);

    // ================= block 2 (shifted) =================
    ln_kernel<2><<<MTOT / 8, 256, 0, stream>>>(t, P[1][0], P[1][1], xw);
    gemm_kernel<0, 0><<<gQKV, blkT, 0, stream>>>(xw, wtQ[1], P[1][3], MTOT, 960, 320,
                                                 qb, kb, vb, nullptr);
    attn_kernel<<<(MTOT / NTOK * HEADS) / 8, 256, 0, stream>>>(qb, kb, vb, biasNN[1], wid, xw, 1);
    gemm_kernel<1, 1><<<gPRJ, blkT, 0, stream>>>(xw, wtP[1], P[1][6], MTOT, 320, 320,
                                                 nullptr, nullptr, nullptr, t);
    ln_kernel<0><<<MTOT / 8, 256, 0, stream>>>(t, P[1][7], P[1][8], qb);
    gemm_kernel<2, 0><<<gF1, blkT, 0, stream>>>(qb, wtF1[1], P[1][10], MTOT, 1280, 320,
                                                hidden, nullptr, nullptr, nullptr);
    gemm_kernel<3, 0><<<gF2, blkT, 0, stream>>>(hidden, wtF2[1], P[1][12], MTOT, 320, 1280,
                                                nullptr, nullptr, nullptr, t);

    // ---- output ----
    transpose_kernel<1><<<BB * 10 * 512, 256, 0, stream>>>(t, (float*)d_out);
}
